// PQLinear_46007689674846
// MI455X (gfx1250) — matrix operand order repaired and checked by reading, not compiled, not run on hardware
//
#include <hip/hip_runtime.h>
#include <hip/hip_bf16.h>
#include <cstdint>

// ---------------------------------------------------------------------------
// PQ linear:  out[M,N] = x[M,K] * W[N,K]^T + bias,  W decoded from centroids.
// M = 8192, N = 4096, K = 4096. Compute-bound (275 GFLOP vs ~0.3GB traffic)
// -> bf16 WMMA path, async-to-LDS double-buffered pipeline.
// ---------------------------------------------------------------------------

typedef __bf16 bf16;
typedef __attribute__((ext_vector_type(16))) __bf16 v16bf;
typedef __attribute__((ext_vector_type(8)))  float  v8f;
typedef __attribute__((ext_vector_type(4)))  int    v4i;

static constexpr int M_TOK   = 4 * 2048;   // 8192
static constexpr int N_OUT   = 4096;
static constexpr int K_IN    = 4096;
static constexpr int NBLOCKS = 512;        // K_IN / 8

static constexpr int BM = 128;
static constexpr int BN = 128;
static constexpr int BK = 32;
static constexpr int LDS_STRIDE = BK + 8;  // 80B rows: 20-bank step, conflict-free

// ---- CDNA5 async global->LDS copy (ASYNCcnt path), guarded for portability --
#if defined(__has_builtin)
#if __has_builtin(__builtin_amdgcn_global_load_async_to_lds_b128)
#define HAVE_ASYNC_LDS 1
#endif
#endif

#define AS1 __attribute__((address_space(1)))
#define AS3 __attribute__((address_space(3)))

__device__ __forceinline__ void copy_b128_to_lds(void* lds_dst, const void* gsrc) {
#if defined(HAVE_ASYNC_LDS)
    // signature (from clang diagnostic): (AS1 int4*, AS3 int4*, imm offset, cpol)
    __builtin_amdgcn_global_load_async_to_lds_b128(
        (AS1 v4i*)(uintptr_t)gsrc, (AS3 v4i*)(uintptr_t)lds_dst, 0, 0);
#else
    *(uint4*)lds_dst = *(const uint4*)gsrc;
#endif
}

__device__ __forceinline__ void wait_async_copies() {
#if defined(HAVE_ASYNC_LDS)
#if __has_builtin(__builtin_amdgcn_s_wait_asynccnt)
    __builtin_amdgcn_s_wait_asynccnt(0);
#else
    asm volatile("s_wait_asynccnt 0x0" ::: "memory");
#endif
#endif
}

// ---------------------------------------------------------------------------
// Kernel 1: decode centroids[assignments] -> dense bf16 weight  W[n][k]
//   assignments laid out [NBLOCKS, N_OUT]; W[n][blk*8+j] = C[a[blk,n]][j]
// ---------------------------------------------------------------------------
__global__ __launch_bounds__(256) void pq_decode_w_kernel(
    const float* __restrict__ centroids,   // [256][8]
    const int*   __restrict__ assign,      // [NBLOCKS * N_OUT]
    bf16*        __restrict__ w)           // [N_OUT][K_IN]
{
    int idx = blockIdx.x * 256 + threadIdx.x;      // 512*4096 threads
    int n   = idx >> 9;                            // / NBLOCKS
    int blk = idx & (NBLOCKS - 1);
    if (n >= N_OUT) return;

    int a = assign[(size_t)blk * N_OUT + n];
    const float4* c = (const float4*)(centroids + (size_t)a * 8);
    float4 c0 = c[0];
    float4 c1 = c[1];

    union { bf16 h[8]; uint4 u; } pk;
    pk.h[0] = (bf16)c0.x; pk.h[1] = (bf16)c0.y;
    pk.h[2] = (bf16)c0.z; pk.h[3] = (bf16)c0.w;
    pk.h[4] = (bf16)c1.x; pk.h[5] = (bf16)c1.y;
    pk.h[6] = (bf16)c1.z; pk.h[7] = (bf16)c1.w;

    *(uint4*)&w[(size_t)n * K_IN + (size_t)blk * 8] = pk.u;
}

// ---------------------------------------------------------------------------
// Kernel 2: convert x f32 -> bf16 (8 elements / thread, 16B store)
// ---------------------------------------------------------------------------
__global__ __launch_bounds__(256) void pq_cvt_x_kernel(
    const float* __restrict__ x, bf16* __restrict__ xb)
{
    size_t i = ((size_t)blockIdx.x * 256 + threadIdx.x) * 8;
    const float4* p = (const float4*)(x + i);
    float4 a = p[0];
    float4 b = p[1];
    union { bf16 h[8]; uint4 u; } pk;
    pk.h[0] = (bf16)a.x; pk.h[1] = (bf16)a.y;
    pk.h[2] = (bf16)a.z; pk.h[3] = (bf16)a.w;
    pk.h[4] = (bf16)b.x; pk.h[5] = (bf16)b.y;
    pk.h[6] = (bf16)b.z; pk.h[7] = (bf16)b.w;
    *(uint4*)&xb[i] = pk.u;
}

// ---------------------------------------------------------------------------
// Kernel 3: WMMA GEMM.  out[M][N] = A[M][K] * B[N][K]^T + bias
//   256 threads = 8 waves (wave32); wave grid 2(M) x 4(N);
//   wave tile 64x32 = 4x2 accumulators of 16x16 f32.
//   Double-buffered LDS, async global->LDS staging overlapped with WMMA.
// Fragment layouts per CDNA5 ISA 7.12.2:
//   A 16x32 bf16: lane<16 -> row M=lane, K {0..7,16..23}; lane>=16 -> K {8..15,24..31}
//   B 32x16 bf16: lane<16 -> col N=lane, K {0..7,16..23}; lane>=16 -> K {8..15,24..31}
// ---------------------------------------------------------------------------
__global__ __launch_bounds__(256) void pq_wmma_gemm_kernel(
    const bf16*  __restrict__ A,     // [M][K] bf16
    const bf16*  __restrict__ Bw,    // [N][K] bf16
    const float* __restrict__ bias,  // [N]
    float*       __restrict__ out)   // [M][N]
{
    __shared__ bf16 As[2][BM][LDS_STRIDE];   // 2 x 10240 B
    __shared__ bf16 Bs[2][BN][LDS_STRIDE];   // 2 x 10240 B

    const int tid  = threadIdx.x;
    const int lane = tid & 31;
    const int wave = tid >> 5;
    const int wm0  = (wave & 1) * 64;   // wave row origin in tile
    const int wn0  = (wave >> 1) * 32;  // wave col origin in tile

    const int m0 = blockIdx.y * BM;
    const int n0 = blockIdx.x * BN;

    const int lrow  = lane & 15;
    const int lhi   = lane >> 4;
    const int koffA = lhi * 8;    // A fragment chunk base (K 0..7 / 8..15)
    const int koffB = lhi * 8;    // B fragment chunk base (K 0..7 / 8..15)

    // staging coordinates: 512 16B-chunks per panel, 2 per thread per panel
    const int srow0 = tid >> 2;            // chunk c = tid      -> rows 0..63
    const int scol  = (tid & 3) * 8;       // 4 chunks of 8 elems per row
    const int srow1 = srow0 + 64;          // chunk c = tid+256  -> rows 64..127

    v8f acc[4][2] = {};

    // ---- prologue: stage panel kb=0 into buffer 0 ----
    copy_b128_to_lds(&As[0][srow0][scol], &A [(size_t)(m0 + srow0) * K_IN + scol]);
    copy_b128_to_lds(&Bs[0][srow0][scol], &Bw[(size_t)(n0 + srow0) * K_IN + scol]);
    copy_b128_to_lds(&As[0][srow1][scol], &A [(size_t)(m0 + srow1) * K_IN + scol]);
    copy_b128_to_lds(&Bs[0][srow1][scol], &Bw[(size_t)(n0 + srow1) * K_IN + scol]);
    wait_async_copies();
    __syncthreads();

    int buf = 0;
    for (int kb = 0; kb < K_IN; kb += BK) {
        // ---- async-stage next panel into the other buffer ----
        if (kb + BK < K_IN) {
            int nb = buf ^ 1;
            int kn = kb + BK;
            copy_b128_to_lds(&As[nb][srow0][scol], &A [(size_t)(m0 + srow0) * K_IN + kn + scol]);
            copy_b128_to_lds(&Bs[nb][srow0][scol], &Bw[(size_t)(n0 + srow0) * K_IN + kn + scol]);
            copy_b128_to_lds(&As[nb][srow1][scol], &A [(size_t)(m0 + srow1) * K_IN + kn + scol]);
            copy_b128_to_lds(&Bs[nb][srow1][scol], &Bw[(size_t)(n0 + srow1) * K_IN + kn + scol]);
        }

        // ---- compute on current buffer (one 32-deep K step, 8 WMMAs/wave) ----
        union { uint4 u[2]; v16bf v; } afrag[4], bfrag[2];
#pragma unroll
        for (int mt = 0; mt < 4; ++mt) {
            const bf16* p = &As[buf][wm0 + mt * 16 + lrow][koffA];
            afrag[mt].u[0] = *(const uint4*)p;         // K +0..7
            afrag[mt].u[1] = *(const uint4*)(p + 16);  // K +16..23
        }
#pragma unroll
        for (int nt = 0; nt < 2; ++nt) {
            const bf16* p = &Bs[buf][wn0 + nt * 16 + lrow][koffB];
            bfrag[nt].u[0] = *(const uint4*)p;         // K +0..7
            bfrag[nt].u[1] = *(const uint4*)(p + 16);  // K +16..23
        }
#pragma unroll
        for (int mt = 0; mt < 4; ++mt)
#pragma unroll
            for (int nt = 0; nt < 2; ++nt)
                acc[mt][nt] = __builtin_amdgcn_wmma_f32_16x16x32_bf16(
                    false, afrag[mt].v, false, bfrag[nt].v,
                    (short)0, acc[mt][nt], false, false);

        // next buffer must be fully staged (and this buffer fully read) before swap
        wait_async_copies();
        __syncthreads();
        buf ^= 1;
    }

    // ---- epilogue: add bias, scatter C fragments ----
    // C 16x16 f32: VGPR r, lanes 0-15 -> M=r, lanes 16-31 -> M=8+r; N=lane&15
#pragma unroll
    for (int nt = 0; nt < 2; ++nt) {
        int gn = n0 + wn0 + nt * 16 + lrow;
        float bv = bias[gn];
#pragma unroll
        for (int mt = 0; mt < 4; ++mt) {
            int gmbase = m0 + wm0 + mt * 16 + lhi * 8;
#pragma unroll
            for (int r = 0; r < 8; ++r) {
                out[(size_t)(gmbase + r) * N_OUT + gn] = acc[mt][nt][r] + bv;
            }
        }
    }
}

// ---------------------------------------------------------------------------
// host launch
// ---------------------------------------------------------------------------
extern "C" void kernel_launch(void* const* d_in, const int* in_sizes, int n_in,
                              void* d_out, int out_size, void* d_ws, size_t ws_size,
                              hipStream_t stream) {
    const float* x         = (const float*)d_in[0];   // [4,2048,4096]
    const float* centroids = (const float*)d_in[1];   // [256,8]
    const int*   assign    = (const int*)  d_in[2];   // [512*4096]
    const float* bias      = (const float*)d_in[3];   // [4096]
    float*       out       = (float*)d_out;

    // workspace: xb (M*K bf16 = 64MB) then wb (N*K bf16 = 32MB)
    bf16* xb = (bf16*)d_ws;
    bf16* wb = (bf16*)((char*)d_ws + (size_t)M_TOK * K_IN * sizeof(bf16));

    // 1) decode weights
    {
        int total = NBLOCKS * N_OUT;                  // 2,097,152
        pq_decode_w_kernel<<<total / 256, 256, 0, stream>>>(centroids, assign, wb);
    }
    // 2) convert x to bf16
    {
        size_t total8 = (size_t)M_TOK * K_IN / 8;     // 4,194,304
        pq_cvt_x_kernel<<<(int)(total8 / 256), 256, 0, stream>>>(x, xb);
    }
    // 3) GEMM with fused bias
    {
        dim3 grid(N_OUT / BN, M_TOK / BM);            // 32 x 64
        pq_wmma_gemm_kernel<<<grid, 256, 0, stream>>>(xb, wb, bias, out);
    }
    (void)in_sizes; (void)n_in; (void)out_size; (void)ws_size;
}
